// MoEFFN_43808666419322
// MI455X (gfx1250) — compile-verified, matrix operand order checked by reading, not run on hardware
//
#include <hip/hip_runtime.h>
#include <hip/hip_bf16.h>
#include <math.h>

typedef __attribute__((ext_vector_type(16))) __bf16 v16bf;
typedef __attribute__((ext_vector_type(8)))  __bf16 v8bf;
typedef __attribute__((ext_vector_type(8)))  float  v8f;

#define NTOK 4096
#define DIM  1024
#define HID  4096
#define NEXP 8

// ---- WMMA 16x32 bf16 fragment load (CDNA5 ISA 7.12.2 layout) ---------------
// Operand rows/cols are contiguous 16-bit data at p; lane-group 0 takes
// elements [0..7, 16..23], group 1 takes [8..15, 24..31] -> two 16B loads.
__device__ __forceinline__ v16bf load_frag(const __bf16* p, int lg) {
  v8bf c0 = *(const v8bf*)(p + 8 * lg);
  v8bf c1 = *(const v8bf*)(p + 8 * lg + 16);
  v16bf r;
#pragma unroll
  for (int j = 0; j < 8; ++j) { r[j] = c0[j]; r[j + 8] = c1[j]; }
  return r;
}

__device__ __forceinline__ v8f wmma_bf16(v16bf a, v16bf b, v8f c) {
  return __builtin_amdgcn_wmma_f32_16x16x32_bf16(false, a, false, b,
                                                 (short)0, c, false, false);
}

// ---- x -> bf16 downconvert --------------------------------------------------
__global__ void cvt_bf16_kernel(const float* __restrict__ x, __bf16* __restrict__ xb, int n) {
  int i = blockIdx.x * blockDim.x + threadIdx.x;
  if (i < n) xb[i] = (__bf16)x[i];
}

// ---- per-expert transpose + downconvert: src [E][R][C] f32 -> dst [E][C][R] bf16
__global__ void transpose_cvt_kernel(const float* __restrict__ src, __bf16* __restrict__ dst,
                                     int R, int C) {
  __shared__ float t[32][33];
  const size_t eoff = (size_t)blockIdx.z * R * C;
  const int c0 = blockIdx.x * 32;
  const int r0 = blockIdx.y * 32;
  const int tx = threadIdx.x;        // 0..31
  const int ty = threadIdx.y;        // 0..7
#pragma unroll
  for (int i = 0; i < 32; i += 8)    // coalesced read along C
    t[ty + i][tx] = src[eoff + (size_t)(r0 + ty + i) * C + (c0 + tx)];
  __syncthreads();
#pragma unroll
  for (int i = 0; i < 32; i += 8)    // coalesced write along R
    dst[eoff + (size_t)(c0 + ty + i) * R + (r0 + tx)] = (__bf16)t[tx][ty + i];
}

// ---- router: logits -> top2 -> renormalized weights -> per-expert scatter ---
__global__ void moe_router_kernel(const float* __restrict__ x, const float* __restrict__ gw,
                                  int* __restrict__ counts, int* __restrict__ list,
                                  float* __restrict__ wlist) {
  const int n    = blockIdx.x * 8 + (threadIdx.x >> 5);  // one wave32 per token
  const int lane = threadIdx.x & 31;
  if (n >= NTOK) return;
  float acc[NEXP];
#pragma unroll
  for (int e = 0; e < NEXP; ++e) acc[e] = 0.f;
  const float* xr = x + (size_t)n * DIM;
  for (int d = lane; d < DIM; d += 32) {
    const float xv = xr[d];
    const float* g = gw + (size_t)d * NEXP;
#pragma unroll
    for (int e = 0; e < NEXP; ++e) acc[e] += xv * g[e];
  }
#pragma unroll
  for (int off = 16; off >= 1; off >>= 1) {
#pragma unroll
    for (int e = 0; e < NEXP; ++e) acc[e] += __shfl_xor(acc[e], off, 32);
  }
  if (lane == 0) {
    int i1 = 0;
    for (int e = 1; e < NEXP; ++e) if (acc[e] > acc[i1]) i1 = e;       // first-max tie rule
    int i2 = (i1 == 0) ? 1 : 0;
    for (int e = 0; e < NEXP; ++e) if (e != i1 && acc[e] > acc[i2]) i2 = e;
    // softmax -> top2 -> renormalize == exp ratio among top-2 logits
    const float m   = fmaxf(acc[i1], acc[i2]);
    const float e1  = expf(acc[i1] - m);
    const float e2  = expf(acc[i2] - m);
    const float inv = 1.0f / (e1 + e2);
    int p1 = atomicAdd(&counts[i1], 1);
    list[i1 * NTOK + p1]  = n;
    wlist[i1 * NTOK + p1] = e1 * inv;
    int p2 = atomicAdd(&counts[i2], 1);
    list[i2 * NTOK + p2]  = n;
    wlist[i2 * NTOK + p2] = e2 * inv;
  }
}

// ---- expert FFN: 64-token tile x one expert, bf16 WMMA, f32 accumulate ------
// Block = 512 threads = 16 waves.  Per chunk of 128 hidden cols:
//   Phase A: wave (w&7 = col tile, w>>3 = row-tile pair) computes h[64x128],
//            GELU+b1 staged to LDS as bf16.
//   Phase B: wave w owns output cols [w*64, +64); accumulates 4x4 C tiles.
__global__ __launch_bounds__(512, 1) void moe_ffn_kernel(
    const __bf16* __restrict__ xb, const __bf16* __restrict__ w1t,
    const float* __restrict__ b1, const __bf16* __restrict__ w2t,
    const float* __restrict__ b2, const int* __restrict__ counts,
    const int* __restrict__ list, const float* __restrict__ wlist,
    float* __restrict__ out) {
  __shared__ __bf16 sh[64][128];    // 16 KB GELU(h) chunk
  __shared__ int    stok[64];
  __shared__ float  swt[64];
  __shared__ int    snv;

  const int e     = blockIdx.y;
  const int tile  = blockIdx.x;
  const int count = counts[e];
  if (tile * 64 >= count) return;                 // uniform early-exit

  const int tid = threadIdx.x;
  if (tid == 0) snv = min(64, count - tile * 64);
  if (tid < 64) {
    const int slot  = tile * 64 + tid;
    const int cslot = slot < count ? slot : count - 1;   // pads alias a real token
    stok[tid] = list[e * NTOK + cslot];
    swt[tid]  = (slot < count) ? wlist[e * NTOK + cslot] : 0.0f;
  }
  __syncthreads();

  const int wv   = tid >> 5;        // 0..15
  const int lane = tid & 31;
  const int lg   = lane >> 4;
  const int ln   = lane & 15;

  const __bf16* w1te = w1t + (size_t)e * HID * DIM;   // [H][D] bf16 (transposed)
  const __bf16* w2te = w2t + (size_t)e * DIM * HID;   // [D][H] bf16 (transposed)
  const float*  b1e  = b1 + e * HID;
  const float*  b2e  = b2 + e * DIM;

  // Phase A assignment
  const int cta = wv & 7;                    // h col tile within chunk
  const int rga = (wv >> 3) * 2;             // first of two row tiles
  const __bf16* pxa0 = xb + (size_t)stok[(rga + 0) * 16 + ln] * DIM;
  const __bf16* pxa1 = xb + (size_t)stok[(rga + 1) * 16 + ln] * DIM;

  // Phase B: per-lane w2t row pointers (fixed output column, contiguous K)
  const __bf16* pw2[4];
#pragma unroll
  for (int ct = 0; ct < 4; ++ct)
    pw2[ct] = w2te + (size_t)(wv * 64 + ct * 16 + ln) * HID;

  v8f yacc[4][4];
#pragma unroll
  for (int rt = 0; rt < 4; ++rt)
#pragma unroll
    for (int ct = 0; ct < 4; ++ct)
#pragma unroll
      for (int i = 0; i < 8; ++i) yacc[rt][ct][i] = 0.f;

  for (int hc = 0; hc < HID / 128; ++hc) {
    // ---------------- Phase A ----------------
    const int hcol = hc * 128 + cta * 16 + ln;
    const __bf16* pw1 = w1te + (size_t)hcol * DIM;   // contiguous K row
    if (hc + 1 < HID / 128) {                        // gfx1250 global_prefetch_b8
      __builtin_prefetch(pw1 + 128 * DIM, 0, 1);
      __builtin_prefetch(pw2[0] + (hc + 1) * 128, 0, 1);
    }
    v8f hacc0, hacc1;
#pragma unroll
    for (int i = 0; i < 8; ++i) { hacc0[i] = 0.f; hacc1[i] = 0.f; }
    for (int kk = 0; kk < DIM; kk += 32) {
      v16bf b  = load_frag(pw1 + kk, lg);            // shared across both row tiles
      v16bf a0 = load_frag(pxa0 + kk, lg);
      v16bf a1 = load_frag(pxa1 + kk, lg);
      hacc0 = wmma_bf16(a0, b, hacc0);
      hacc1 = wmma_bf16(a1, b, hacc1);
    }
    const float b1v = b1e[hcol];
    __syncthreads();                                 // prior Phase B done with sh
#pragma unroll
    for (int i = 0; i < 8; ++i) {                    // C layout: VGPR i -> row i + 8*lg
      const float v0 = hacc0[i] + b1v;
      const float v1 = hacc1[i] + b1v;
      sh[(rga + 0) * 16 + i + 8 * lg][cta * 16 + ln] =
          (__bf16)(0.5f * v0 * (1.0f + erff(v0 * 0.70710678118654752f)));
      sh[(rga + 1) * 16 + i + 8 * lg][cta * 16 + ln] =
          (__bf16)(0.5f * v1 * (1.0f + erff(v1 * 0.70710678118654752f)));
    }
    __syncthreads();                                 // sh chunk ready

    // ---------------- Phase B ----------------
#pragma unroll
    for (int kk2 = 0; kk2 < 128; kk2 += 32) {
      v16bf af[4];
#pragma unroll
      for (int rt = 0; rt < 4; ++rt)
        af[rt] = load_frag(&sh[rt * 16 + ln][kk2], lg);
#pragma unroll
      for (int ct = 0; ct < 4; ++ct) {
        v16bf b = load_frag(pw2[ct] + hc * 128 + kk2, lg);
#pragma unroll
        for (int rt = 0; rt < 4; ++rt)
          yacc[rt][ct] = wmma_bf16(af[rt], b, yacc[rt][ct]);
      }
    }
  }

  // ---- combine: out[token] += w_tok * (y + b2) (token can be in 2 experts) ----
  const int nv = snv;
#pragma unroll
  for (int ct = 0; ct < 4; ++ct) {
    const int dcol  = wv * 64 + ct * 16 + ln;
    const float b2v = b2e[dcol];
#pragma unroll
    for (int rt = 0; rt < 4; ++rt) {
#pragma unroll
      for (int i = 0; i < 8; ++i) {
        const int m = rt * 16 + i + 8 * lg;
        if (m < nv)
          atomicAdd(&out[(size_t)stok[m] * DIM + dcol], swt[m] * (yacc[rt][ct][i] + b2v));
      }
    }
  }
}

// ---- host launcher ----------------------------------------------------------
extern "C" void kernel_launch(void* const* d_in, const int* in_sizes, int n_in,
                              void* d_out, int out_size, void* d_ws, size_t ws_size,
                              hipStream_t stream) {
  const float* x  = (const float*)d_in[0];
  const float* gw = (const float*)d_in[1];
  const float* w1 = (const float*)d_in[2];
  const float* b1 = (const float*)d_in[3];
  const float* w2 = (const float*)d_in[4];
  const float* b2 = (const float*)d_in[5];
  float* out = (float*)d_out;

  // ws layout: counts | list | wlist | xb | w1t | w2t   (~136.3 MB)
  char* ws = (char*)d_ws;
  int*    counts = (int*)ws;                                             // 64 B
  int*    list   = (int*)(ws + 64);                                      // E*N ints
  float*  wlist  = (float*)(ws + 64 + sizeof(int) * NEXP * NTOK);        // E*N floats
  __bf16* xb     = (__bf16*)(ws + 64 + 8 * NEXP * NTOK);                 // N*D bf16
  __bf16* w1t    = xb + (size_t)NTOK * DIM;                              // E*H*D bf16
  __bf16* w2t    = w1t + (size_t)NEXP * HID * DIM;                       // E*D*H bf16

  hipMemsetAsync(counts, 0, 64, stream);
  hipMemsetAsync(d_out, 0, sizeof(float) * (size_t)out_size, stream);

  cvt_bf16_kernel<<<(NTOK * DIM + 255) / 256, 256, 0, stream>>>(x, xb, NTOK * DIM);
  // w1 [E][D][H] -> w1t [E][H][D] ; w2 [E][H][D_out] -> w2t [E][D_out][H]
  transpose_cvt_kernel<<<dim3(HID / 32, DIM / 32, NEXP), dim3(32, 8), 0, stream>>>(w1, w1t, DIM, HID);
  transpose_cvt_kernel<<<dim3(DIM / 32, HID / 32, NEXP), dim3(32, 8), 0, stream>>>(w2, w2t, HID, DIM);
  moe_router_kernel<<<NTOK / 8, 256, 0, stream>>>(x, gw, counts, list, wlist);
  moe_ffn_kernel<<<dim3(NTOK / 64, NEXP), 512, 0, stream>>>(
      xb, w1t, b1, w2t, b2, counts, list, wlist, out);
}
